// PosteriorSDE_37555194036846
// MI455X (gfx1250) — matrix operand order repaired and checked by reading, not compiled, not run on hardware
//
#include <hip/hip_runtime.h>
#include <cmath>

typedef __attribute__((ext_vector_type(16))) _Float16 v16h;
typedef __attribute__((ext_vector_type(8)))  _Float16 v8h;
typedef __attribute__((ext_vector_type(8)))  float    v8f;

#define B_   256
#define T_   256
#define D_   128
#define E_   32
#define F1_  512
#define F2_  512
#define KIN_ (D_ + E_)          // 160
#define KC1_ 5                  // 160/32 K-chunks for GEMM1
#define KC2_ 16                 // 512/32 K-chunks for GEMM2/3
#define ALPHA_   0.1f
#define SCALE_   0.31622776601683794f   // sqrt(0.1)
#define ROWS_    520            // padded LDS row stride in halfs (16B aligned rows)

#if __has_builtin(__builtin_amdgcn_tanhf)
#define TANH_(x) __builtin_amdgcn_tanhf(x)   // native v_tanh_f32 on gfx1250
#else
#define TANH_(x) tanhf(x)
#endif

// ---------------------------------------------------------------------------
// Weight converter: fp32 row-major [K,N]  ->  f16 fragment-major layout.
// Fragment (nt, kc) = 32x16 block of W; 512 halfs, lane-major:
//   out[frag*512 + lane*16 + j] = W[kc*32 + (j/8)*16 + (lane/16)*8 + j%8][nt*16 + lane%16]
// so each WMMA lane's B operand is 16 contiguous halfs (32 bytes).
// ---------------------------------------------------------------------------
__global__ void frag_convert(const float* __restrict__ W, _Float16* __restrict__ out,
                             int K, int N, int total) {
  int idx = blockIdx.x * blockDim.x + threadIdx.x;
  if (idx >= total) return;
  int frag = idx >> 9;
  int rem  = idx & 511;
  int lane = rem >> 4;
  int j    = rem & 15;
  int nkc  = (K + 31) >> 5;
  int nt   = frag / nkc;
  int kc   = frag - nt * nkc;
  int k    = kc * 32 + (j >> 3) * 16 + (lane >> 4) * 8 + (j & 7);
  int n    = nt * 16 + (lane & 15);
  float v  = (k < K) ? W[(size_t)k * N + n] : 0.0f;
  out[idx] = (_Float16)v;
}

// A fragment (16x32 f16, documented CDNA5 layout): lane L holds row m=L%16;
// halves 0..7 = K in [8*(L/16), +8), halves 8..15 = K in [16+8*(L/16), +8).
// Row-major LDS => two contiguous 16-byte ds_load_b128s.
__device__ __forceinline__ v16h load_afrag(const _Float16* __restrict__ buf,
                                           int kc, int lane) {
  union { v16h v; v8h h[2]; } u;
  const int m  = lane & 15;
  const int hi = (lane >> 4) * 8;
  const _Float16* p = buf + m * ROWS_ + kc * 32 + hi;
  u.h[0] = *(const v8h*)(p);
  u.h[1] = *(const v8h*)(p + 16);
  return u.v;
}

__device__ __forceinline__ v16h load_bfrag(const _Float16* __restrict__ Wh,
                                           int nt, int nkc, int kc, int lane) {
  return *(const v16h*)(Wh + (((size_t)nt * nkc + kc) << 9) + lane * 16);
}

// C/D layout: lane L holds column n=L%16, rows m = r + 8*(L/16), r=0..7.
// Apply bias + tanh (native), convert to f16, scatter into row-major LDS.
__device__ __forceinline__ void store_tile_tanh(v8f acc, _Float16* __restrict__ dst,
                                                int nt, const float* __restrict__ bias,
                                                int lane) {
  const int n     = nt * 16 + (lane & 15);
  const int mbase = (lane >> 4) * 8;
  const float bv  = bias[n];
#pragma unroll
  for (int r = 0; r < 8; ++r) {
    float v = TANH_(acc[r] + bv);
    dst[(mbase + r) * ROWS_ + n] = (_Float16)v;
  }
}

__global__ void __launch_bounds__(256, 2)
sde_scan(const float* __restrict__ carry, const float* __restrict__ x,
         const float* __restrict__ ext,   const float* __restrict__ noise,
         const _Float16* __restrict__ W1h, const float* __restrict__ gb1,
         const _Float16* __restrict__ W2h, const float* __restrict__ gb2,
         const _Float16* __restrict__ W3h, const float* __restrict__ gb3,
         float* __restrict__ out) {
  __shared__ _Float16 hb0[16 * ROWS_];   // A0 / h2 ping-pong
  __shared__ _Float16 hb1[16 * ROWS_];   // h1
  __shared__ float    yf[16 * D_];       // current state y (fp32)
  __shared__ float    sb1[F1_], sb2[F2_], sb3[D_];

  const int tid  = threadIdx.x;
  const int lane = tid & 31;
  const int w    = tid >> 5;             // wave id 0..7
  const int row0 = blockIdx.x * 16;      // batch-row tile

  for (int i = tid; i < F1_; i += 256) sb1[i] = gb1[i];
  for (int i = tid; i < F2_; i += 256) sb2[i] = gb2[i];
  for (int i = tid; i < D_;  i += 256) sb3[i] = gb3[i];
  for (int i = tid; i < 16 * D_; i += 256) {
    int m = i >> 7, c = i & (D_ - 1);
    yf[i] = carry[(size_t)(row0 + m) * D_ + c];
  }
  __syncthreads();

  float* out_yfin = out;
  float* out_ys   = out + (size_t)B_ * D_;
  float* out_mp   = out_ys + (size_t)B_ * T_ * D_;
  float* out_mu   = out_mp + (size_t)B_ * T_ * D_;
  float* out_sd   = out_mu + (size_t)B_ * T_ * D_;

  for (int t = 0; t < T_; ++t) {
    // ---- stage A0 = concat(y, u_t) as f16 in hb0 ------------------------
    for (int i = tid; i < 16 * KIN_; i += 256) {
      int m = i / KIN_, c = i - m * KIN_;
      float v;
      if (c < D_) v = yf[m * D_ + c];
      else        v = ext[((size_t)(row0 + m) * T_ + t) * E_ + (c - D_)];
      hb0[m * ROWS_ + c] = (_Float16)v;
    }
    __syncthreads();

    // ---- GEMM1: h1 = tanh(A0 @ W1 + b1)   [16 x 512], K=160 ------------
#pragma unroll 1
    for (int it = 0; it < 4; ++it) {
      const int nt = w + it * 8;
      v8f acc = {};
#pragma unroll 5
      for (int kc = 0; kc < KC1_; ++kc) {
        v16h a = load_afrag(hb0, kc, lane);
        v16h b = load_bfrag(W1h, nt, KC1_, kc, lane);
        acc = __builtin_amdgcn_wmma_f32_16x16x32_f16(false, a, false, b,
                                                     (short)0, acc, false, false);
      }
      store_tile_tanh(acc, hb1, nt, sb1, lane);
    }
    __syncthreads();

    // ---- GEMM2: h2 = tanh(h1 @ W2 + b2)   [16 x 512], K=512 ------------
#pragma unroll 1
    for (int it = 0; it < 4; ++it) {
      const int nt = w + it * 8;
      v8f acc = {};
#pragma unroll 4
      for (int kc = 0; kc < KC2_; ++kc) {
        v16h a = load_afrag(hb1, kc, lane);
        v16h b = load_bfrag(W2h, nt, KC2_, kc, lane);
        acc = __builtin_amdgcn_wmma_f32_16x16x32_f16(false, a, false, b,
                                                     (short)0, acc, false, false);
      }
      store_tile_tanh(acc, hb0, nt, sb2, lane);
    }
    __syncthreads();

    // ---- GEMM3: mu_phi = h2 @ W3 + b3     [16 x 128], K=512 ------------
    {
      const int nt = w;                       // 8 waves x 16 cols = 128
      v8f acc = {};
#pragma unroll 4
      for (int kc = 0; kc < KC2_; ++kc) {
        v16h a = load_afrag(hb0, kc, lane);
        v16h b = load_bfrag(W3h, nt, KC2_, kc, lane);
        acc = __builtin_amdgcn_wmma_f32_16x16x32_f16(false, a, false, b,
                                                     (short)0, acc, false, false);
      }
      // ---- elementwise SDE update straight from the C fragment ----------
      const int n     = lane & 15;
      const int dcol  = nt * 16 + n;
      const int mbase = (lane >> 4) * 8;
      const float b3v = sb3[dcol];

      if (t + 1 < T_)   // pull next step's x row toward L2/L0
        __builtin_prefetch(&x[((size_t)(row0 + mbase) * T_ + (t + 1)) * (2 * D_) + dcol], 0, 0);

#pragma unroll
      for (int r = 0; r < 8; ++r) {
        const int m = mbase + r;
        const int brow = row0 + m;
        const size_t xbase = ((size_t)brow * T_ + t) * (2 * D_);
        const size_t obase = ((size_t)brow * T_ + t) * D_ + dcol;

        float mu_phi = acc[r] + b3v;
        float yold   = yf[m * D_ + dcol];
        float x1     = x[xbase + dcol];
        float x2     = x[xbase + D_ + dcol];
        float mu     = (1.0f - ALPHA_) * yold + ALPHA_ * mu_phi + x1;
        float sp     = (x2 > 20.0f) ? x2 : __logf(1.0f + __expf(x2));
        float sd     = SCALE_ * sp;
        float nv     = noise[obase];
        float ynew   = mu + sd * nv;

        out_ys[obase] = ynew;
        out_mp[obase] = mu_phi;
        out_mu[obase] = mu;
        out_sd[obase] = sd;
        yf[m * D_ + dcol] = ynew;
      }
    }
    __syncthreads();   // yf update visible before next A0 build
  }

  // ---- y_final ----------------------------------------------------------
  for (int i = tid; i < 16 * D_; i += 256) {
    int m = i >> 7, c = i & (D_ - 1);
    out_yfin[(size_t)(row0 + m) * D_ + c] = yf[i];
  }
}

extern "C" void kernel_launch(void* const* d_in, const int* in_sizes, int n_in,
                              void* d_out, int out_size, void* d_ws, size_t ws_size,
                              hipStream_t stream) {
  const float* carry = (const float*)d_in[0];
  const float* x     = (const float*)d_in[1];
  const float* ext   = (const float*)d_in[2];
  const float* noise = (const float*)d_in[3];
  const float* W1    = (const float*)d_in[4];
  const float* b1    = (const float*)d_in[5];
  const float* W2    = (const float*)d_in[6];
  const float* b2    = (const float*)d_in[7];
  const float* W3    = (const float*)d_in[8];
  const float* b3    = (const float*)d_in[9];

  // f16 fragment-major weights in workspace (819,200 bytes total)
  const int n1 = KIN_ * F1_;   // 81920
  const int n2 = F1_  * F2_;   // 262144
  const int n3 = F2_  * D_;    // 65536
  _Float16* W1h = (_Float16*)d_ws;
  _Float16* W2h = W1h + n1;
  _Float16* W3h = W2h + n2;

  frag_convert<<<(n1 + 255) / 256, 256, 0, stream>>>(W1, W1h, KIN_, F1_, n1);
  frag_convert<<<(n2 + 255) / 256, 256, 0, stream>>>(W2, W2h, F1_,  F2_, n2);
  frag_convert<<<(n3 + 255) / 256, 256, 0, stream>>>(W3, W3h, F2_,  D_,  n3);

  sde_scan<<<B_ / 16, 256, 0, stream>>>(carry, x, ext, noise,
                                        W1h, b1, W2h, b2, W3h, b3,
                                        (float*)d_out);
}
